// UniMP_70196945486353
// MI455X (gfx1250) — matrix operand order repaired and checked by reading, not compiled, not run on hardware
//
#include <hip/hip_runtime.h>
#include <hip/hip_bf16.h>
#include <math.h>

// ---------------------------------------------------------------------------
// CDNA5 (gfx1250) UniMP GNN forward.
// All dense layers run through v_wmma_f32_16x16x32_bf16 (wave32 WMMA).
// ---------------------------------------------------------------------------

typedef __attribute__((ext_vector_type(16))) __bf16   bf16x16;
typedef __attribute__((ext_vector_type(2)))  __bf16   bf16x2;
typedef __attribute__((ext_vector_type(8)))  float    f32x8;
typedef __attribute__((ext_vector_type(8)))  unsigned int u32x8;

__device__ __forceinline__ unsigned short f2bf(float f) {
    unsigned int u = __float_as_uint(f);
    u += 0x7FFFu + ((u >> 16) & 1u);        // round-to-nearest-even
    return (unsigned short)(u >> 16);
}

// pack two f32 -> one dword of bf16 (v_cvt_pk_bf16_f32 when available)
__device__ __forceinline__ unsigned int f2bf2(float lo, float hi) {
#if __has_builtin(__builtin_amdgcn_cvt_pk_bf16_f32)
    bf16x2 p = __builtin_amdgcn_cvt_pk_bf16_f32(lo, hi);
    return __builtin_bit_cast(unsigned int, p);
#else
    return (unsigned int)f2bf(lo) | ((unsigned int)f2bf(hi) << 16);
#endif
}

__device__ __forceinline__ void atomicMaxF(float* addr, float val) {
    unsigned int* ua = (unsigned int*)addr;
    unsigned int cur = *ua;
    while (__uint_as_float(cur) < val) {
        unsigned int prev = atomicCAS(ua, cur, __float_as_uint(val));
        if (prev == cur) break;
        cur = prev;
    }
}

// ---------------------------------------------------------------------------
// Generic bf16-WMMA GEMM:  C[M x O] = (rowscale .* A[M x K]) @ W  (+ bias)
//  - A row-major with row stride lda (lets us read strided branch buffers)
//  - wsem==0: W is row-major [K x O]
//    wsem==1: SE-module layout W[j, k, d] with j = o/256, d = o%256
//  - Block: 256 threads = 8 wave32; tile 128(M) x 64(O); K step 64;
//    8 v_wmma_f32_16x16x32_bf16 per stage (A fragment reused 4x per half).
// ---------------------------------------------------------------------------
__global__ __launch_bounds__(256) void gemm_bf16_wmma_kernel(
    const float* __restrict__ A, int lda,
    const float* __restrict__ W, int wsem,
    const float* __restrict__ bias, const float* __restrict__ rowscale,
    float* __restrict__ C, int ldc, int M, int K, int O)
{
    __shared__ unsigned short At[128][66];   // 128 x 64 bf16 (+pad)
    __shared__ unsigned short Wt[64][66];    // transposed 64(O) x 64(K) bf16

    const int tid  = threadIdx.x;
    const int lane = tid & 31;
    const int wid  = tid >> 5;
    const int mBase = blockIdx.y * 128;
    const int oBase = blockIdx.x * 64;

    f32x8 zero = {};
    f32x8 acc[4] = { zero, zero, zero, zero };

    // A-tile loader: thread -> (row 0..127, 32-wide K half)
    const int ar  = tid >> 1;
    const int ak0 = (tid & 1) * 32;
    const int gm  = mBase + ar;
    const float rs = (rowscale && gm < M) ? rowscale[gm] : 1.0f;
    const float* arow = A + (size_t)gm * lda;
    // W-tile loader: thread -> (col 0..63, 16 consecutive K)
    const int wo  = tid & 63;
    const int wk0 = (tid >> 6) * 16;
    const int go  = oBase + wo;
    const int wj  = go >> 8;        // SE-module block index
    const int wd  = go & 255;

    const int ksteps = (K + 63) >> 6;
    for (int ks = 0; ks < ksteps; ++ks) {
        const int kb = ks << 6;

        // ---- stage A tile (f32 -> bf16 packed, optional per-row scale) ----
        if (gm < M && kb + 64 < K)
            __builtin_prefetch(&arow[kb + 64 + ak0], 0, 1);
        #pragma unroll
        for (int c = 0; c < 8; ++c) {
            int kk = kb + ak0 + c * 4;
            if (gm < M && kk + 3 < K) {
                float4 f = *(const float4*)(arow + kk);
                *(unsigned int*)&At[ar][ak0 + c * 4]     = f2bf2(f.x * rs, f.y * rs);
                *(unsigned int*)&At[ar][ak0 + c * 4 + 2] = f2bf2(f.z * rs, f.w * rs);
            } else {
                float v0 = (gm < M && kk + 0 < K) ? arow[kk + 0] * rs : 0.0f;
                float v1 = (gm < M && kk + 1 < K) ? arow[kk + 1] * rs : 0.0f;
                float v2 = (gm < M && kk + 2 < K) ? arow[kk + 2] * rs : 0.0f;
                float v3 = (gm < M && kk + 3 < K) ? arow[kk + 3] * rs : 0.0f;
                *(unsigned int*)&At[ar][ak0 + c * 4]     = f2bf2(v0, v1);
                *(unsigned int*)&At[ar][ak0 + c * 4 + 2] = f2bf2(v2, v3);
            }
        }
        // ---- stage W tile transposed (f32 -> bf16 packed) ----
        #pragma unroll
        for (int i = 0; i < 16; i += 2) {
            int k0 = wk0 + i, kk0 = kb + k0;
            float v0 = 0.0f, v1 = 0.0f;
            if (kk0 < K)
                v0 = wsem ? W[((size_t)wj * K + kk0) * 256 + wd]
                          : W[(size_t)kk0 * O + go];
            if (kk0 + 1 < K)
                v1 = wsem ? W[((size_t)wj * K + kk0 + 1) * 256 + wd]
                          : W[(size_t)(kk0 + 1) * O + go];
            *(unsigned int*)&Wt[wo][k0] = f2bf2(v0, v1);
        }
        __syncthreads();

        // ---- two 32-deep WMMA halves per stage ----
        const int mrow  = wid * 16 + (lane & 15);
        const int abase = (lane < 16) ? 0 : 8;
        const int kbB   = (lane < 16) ? 0 : 16;
        const int ncol  = lane & 15;
        #pragma unroll
        for (int h = 0; h < 2; ++h) {
            const int kh = h * 32;
            // A fragment: 16x32 bf16 per wave (ISA 7.12.2 layout)
            u32x8 au;
            #pragma unroll
            for (int v = 0; v < 8; ++v) {
                int kp = kh + ((v < 4) ? (2 * v) : (16 + 2 * (v - 4))) + abase;
                au[v] = *(const unsigned int*)&At[mrow][kp];
            }
            bf16x16 av = __builtin_bit_cast(bf16x16, au);
            // 4 WMMAs across the 64-wide O strip
            #pragma unroll
            for (int j = 0; j < 4; ++j) {
                u32x8 bu;
                #pragma unroll
                for (int v = 0; v < 8; ++v)
                    bu[v] = *(const unsigned int*)&Wt[j * 16 + ncol][kh + ((v < 4) ? (2 * v) : (16 + 2 * (v - 4))) + abase];
                bf16x16 bv = __builtin_bit_cast(bf16x16, bu);
                acc[j] = __builtin_amdgcn_wmma_f32_16x16x32_bf16(
                    false, av, false, bv, (short)0, acc[j], false, false);
            }
        }
        __syncthreads();
    }

    // ---- store C (f32 C/D layout: VGPR r -> M = r (+8 for hi lanes)) ----
    const int colb = oBase + (lane & 15);
    const int mtop = mBase + wid * 16 + ((lane < 16) ? 0 : 8);
    #pragma unroll
    for (int j = 0; j < 4; ++j) {
        int col = colb + j * 16;
        float bv = bias ? bias[col] : 0.0f;
        #pragma unroll
        for (int r = 0; r < 8; ++r) {
            int m = mtop + r;
            if (m < M) C[(size_t)m * ldc + col] = acc[j][r] + bv;
        }
    }
}

// ---------------------------------------------------------------------------
// Wave-per-row LayerNorm (+ optional pre row-scale, pre bias, ELU). D=256.
// ---------------------------------------------------------------------------
__global__ __launch_bounds__(256) void ln_act_kernel(
    const float* __restrict__ X, int ldx,
    const float* __restrict__ rowscale, const float* __restrict__ prebias,
    const float* __restrict__ lnw, const float* __restrict__ lnb,
    float* __restrict__ Y, int ldy, int M, int elu)
{
    int row = blockIdx.x * 8 + (threadIdx.x >> 5);
    if (row >= M) return;
    int lane = threadIdx.x & 31;
    const float rs = rowscale ? rowscale[row] : 1.0f;
    const float* xr = X + (size_t)row * ldx;
    float v[8], s = 0.f, sq = 0.f;
    #pragma unroll
    for (int i = 0; i < 8; ++i) {
        int d = i * 32 + lane;
        float t = xr[d] * rs + (prebias ? prebias[d] : 0.f);
        v[i] = t; s += t; sq += t * t;
    }
    #pragma unroll
    for (int o = 16; o; o >>= 1) {
        s  += __shfl_xor(s,  o, 32);
        sq += __shfl_xor(sq, o, 32);
    }
    float mean = s * (1.f / 256.f);
    float var  = sq * (1.f / 256.f) - mean * mean;
    float rstd = rsqrtf(var + 1e-12f);
    float* yr = Y + (size_t)row * ldy;
    #pragma unroll
    for (int i = 0; i < 8; ++i) {
        int d = i * 32 + lane;
        float t = (v[i] - mean) * rstd * lnw[d] + lnb[d];
        if (elu) t = (t > 0.f) ? t : expm1f(t);
        yr[d] = t;
    }
}

// ---------------------------------------------------------------------------
// Utility / edge kernels
// ---------------------------------------------------------------------------
__global__ void fill_kernel(float* p, float v, long long n) {
    long long i = (long long)blockIdx.x * 256 + threadIdx.x;
    if (i < n) p[i] = v;
}

__global__ void deg_kernel(const int* __restrict__ src, const int* __restrict__ dst,
                           float* dout, float* din, long long E) {
    long long t = (long long)blockIdx.x * 256 + threadIdx.x;
    if (t >= E) return;
    atomicAdd(&dout[src[t]], 1.0f);
    atomicAdd(&din[dst[t]], 1.0f);
}

__global__ void deg_finalize_kernel(float* p, long long n) {
    long long i = (long long)blockIdx.x * 256 + threadIdx.x;
    if (i < n) p[i] = rsqrtf(fmaxf(p[i], 1.0f));
}

// agg[dst] += tmp[src]  (256 feats, float4 per thread)
__global__ void scatter_add_kernel(const float* __restrict__ tmp,
                                   const int* __restrict__ src, const int* __restrict__ dst,
                                   float* __restrict__ agg, long long E) {
    long long t = (long long)blockIdx.x * 256 + threadIdx.x;
    if (t >= E * 64) return;
    long long e = t >> 6;
    int off = (int)(t & 63) * 4;
    int s = src[e], d = dst[e];
    const float4 f = *(const float4*)(tmp + (size_t)s * 256 + off);
    float* po = agg + (size_t)d * 256 + off;
    atomicAdd(po + 0, f.x); atomicAdd(po + 1, f.y);
    atomicAdd(po + 2, f.z); atomicAdd(po + 3, f.w);
}

// GATv2 pass 1: per-edge per-head logits + running max per (dst, head)
__global__ __launch_bounds__(256) void attn_logit_kernel(
    const float* __restrict__ fs, const float* __restrict__ fd,
    const float* __restrict__ a, const int* __restrict__ src,
    const int* __restrict__ dst, float* __restrict__ logit,
    float* __restrict__ mmax, long long E)
{
    long long e = (long long)blockIdx.x * 8 + (threadIdx.x >> 5);
    if (e >= E) return;
    int lane = threadIdx.x & 31;
    int h = lane >> 2, sub = lane & 3;
    int s = src[e], d = dst[e];
    int base = h * 32 + sub * 8;
    const float* ps = fs + (size_t)s * 256 + base;
    const float* pd = fd + (size_t)d * 256 + base;
    const float* pa = a + base;
    float p = 0.f;
    #pragma unroll
    for (int i = 0; i < 8; ++i) {
        float v = ps[i] + pd[i];
        v = (v > 0.f) ? v : 0.2f * v;      // leaky_relu(0.2)
        p += v * pa[i];
    }
    p += __shfl_xor(p, 1, 32);
    p += __shfl_xor(p, 2, 32);
    if (sub == 0) {
        logit[e * 8 + h] = p;
        atomicMaxF(&mmax[(size_t)d * 8 + h], p);
    }
}

// GATv2 pass 2: w = exp(logit - max), denom[dst,h] += w
__global__ void attn_expsum_kernel(const float* __restrict__ logit,
                                   const float* __restrict__ mmax,
                                   const int* __restrict__ dst,
                                   float* __restrict__ wexp,
                                   float* __restrict__ denom, long long E)
{
    long long t = (long long)blockIdx.x * 256 + threadIdx.x;
    if (t >= E * 8) return;
    long long e = t >> 3; int h = (int)(t & 7);
    int d = dst[e];
    float mm = mmax[(size_t)d * 8 + h];
    if (!isfinite(mm)) mm = 0.f;
    float w = expf(logit[t] - mm);
    wexp[t] = w;
    atomicAdd(&denom[(size_t)d * 8 + h], w);
}

// GATv2 pass 3: out[dst] += fs[src] * alpha
__global__ void attn_agg_kernel(const float* __restrict__ fs,
                                const float* __restrict__ wexp,
                                const float* __restrict__ denom,
                                const int* __restrict__ src, const int* __restrict__ dst,
                                float* __restrict__ outp, long long E)
{
    long long t = (long long)blockIdx.x * 256 + threadIdx.x;
    if (t >= E * 64) return;
    long long e = t >> 6;
    int idx = (int)(t & 63);
    int h = idx >> 3, q = idx & 7;
    int s = src[e], d = dst[e];
    float alpha = wexp[e * 8 + h] / denom[(size_t)d * 8 + h];
    int off = h * 32 + q * 4;
    const float4 f = *(const float4*)(fs + (size_t)s * 256 + off);
    float* po = outp + (size_t)d * 256 + off;
    atomicAdd(po + 0, f.x * alpha); atomicAdd(po + 1, f.y * alpha);
    atomicAdd(po + 2, f.z * alpha); atomicAdd(po + 3, f.w * alpha);
}

// SE-module gate: out[n,d] = sum_j feats[n,j,d] * sigmoid(z[n,j,d]*INV*g+bb)
__global__ void semul_gate_kernel(const float* __restrict__ feats,
                                  const float* __restrict__ z, int ldf,
                                  const float* __restrict__ g, const float* __restrict__ bb,
                                  int nb, float* __restrict__ out, int N)
{
    long long t = (long long)blockIdx.x * 256 + threadIdx.x;
    if (t >= (long long)N * 256) return;
    int n = (int)(t >> 8), d = (int)(t & 255);
    const float INV = 0.9999950000374997f;  // 1/sqrt(1+1e-5) (BN eval stats)
    float acc = 0.f;
    for (int j = 0; j < nb; ++j) {
        float f  = feats[(size_t)n * ldf + j * 256 + d];
        float zz = z[(size_t)n * ldf + j * 256 + d];
        float gg = 1.f / (1.f + expf(-(zz * INV * g[j * 256 + d] + bb[j * 256 + d])));
        acc += f * gg;
    }
    out[(size_t)n * 256 + d] = acc;
}

// Output projection 256 -> 23
__global__ void out_proj_kernel(const float* __restrict__ h,
                                const float* __restrict__ W, const float* __restrict__ b,
                                float* __restrict__ out, int N)
{
    long long t = (long long)blockIdx.x * 256 + threadIdx.x;
    if (t >= (long long)N * 23) return;
    int n = (int)(t / 23), c = (int)(t % 23);
    const float* hr = h + (size_t)n * 256;
    float s = b[c];
    #pragma unroll 8
    for (int d = 0; d < 256; ++d) s += hr[d] * W[d * 23 + c];
    out[t] = s;
}

// ---------------------------------------------------------------------------
// Host orchestration.
// Input order: setup_inputs() dict order {x, params, src, dst}; params
// flattened in JAX pytree (sorted-key) order:
//  0:x | attn[0]: 1:Wdst 2:Wsrc 3:a 4:bdst 5:bias 6:bsrc 7:ln_b 8:ln_w
//  attn[1]: 9..16 | conv[0]: 17:W 18:b 19:ln_b 20:ln_w | conv[1]: 21..24
//  25:fln_b 26:fln_w 27:out_W 28:out_b
//  semul[0]: 29:W 30:b 31:bb 32:g | semul[1]: 33..36
//  skip[0][0]: 37..40 | skip[1][0]: 41..44 | skip[1][1]: 45..48
//  49:src 50:dst
// ---------------------------------------------------------------------------
extern "C" void kernel_launch(void* const* d_in, const int* in_sizes, int n_in,
                              void* d_out, int out_size, void* d_ws, size_t ws_size,
                              hipStream_t stream)
{
#define FIN(i) ((const float*)d_in[(i)])
    const float* x = FIN(0);
    const int*  src = (const int*)d_in[49];
    const int*  dst = (const int*)d_in[50];
    const int   N = in_sizes[0] / 300;
    const long long E = in_sizes[49];

    // ---- carve workspace ----
    size_t off = 0;
    auto carve = [&](size_t nfloats) {
        float* p = (float*)((char*)d_ws + off);
        off += ((nfloats * sizeof(float)) + 255) & ~(size_t)255;
        return p;
    };
    float* degs   = carve((size_t)2 * N);          // [deg_out | deg_in] ^-1/2
    float* deg_o  = degs;
    float* deg_i  = degs + N;
    float* feats0 = carve((size_t)N * 768);
    float* z0     = carve((size_t)N * 768);
    float* feats1 = carve((size_t)N * 1024);
    float* z1     = carve((size_t)N * 1024);
    float* tmp    = carve((size_t)N * 256);
    float* agg    = carve((size_t)N * 256);
    float* fs     = carve((size_t)N * 256);
    float* fd     = carve((size_t)N * 256);
    float* logit  = carve((size_t)E * 8);
    float* wexp   = carve((size_t)E * 8);
    float* mmax   = carve((size_t)N * 8);
    float* denom  = carve((size_t)N * 8);
    float* h1     = carve((size_t)N * 256);
    float* hln    = carve((size_t)N * 256);
    (void)ws_size;

    auto fill = [&](float* p, float v, long long n) {
        fill_kernel<<<(unsigned)((n + 255) / 256), 256, 0, stream>>>(p, v, n);
    };
    auto gemm = [&](const float* A, int lda, const float* W, int wsem,
                    const float* bias, const float* rowscale,
                    float* C, int ldc, int M, int K, int O) {
        dim3 g(O / 64, (M + 127) / 128);
        gemm_bf16_wmma_kernel<<<g, 256, 0, stream>>>(A, lda, W, wsem, bias,
                                                     rowscale, C, ldc, M, K, O);
    };
    auto ln = [&](const float* X, int ldx, const float* rs, const float* pb,
                  const float* w, const float* b, float* Y, int ldy, int elu) {
        ln_act_kernel<<<(unsigned)((N + 7) / 8), 256, 0, stream>>>(
            X, ldx, rs, pb, w, b, Y, ldy, N, elu);
    };
    const float NEG_INF = -__builtin_inff();

    // ---- degrees ----
    fill(degs, 0.f, 2LL * N);
    deg_kernel<<<(unsigned)((E + 255) / 256), 256, 0, stream>>>(src, dst, deg_o, deg_i, E);
    deg_finalize_kernel<<<(unsigned)((2LL * N + 255) / 256), 256, 0, stream>>>(degs, 2LL * N);

    // ================= Layer 0 (input x, F=300) =================
    // GraphConv branch -> feats0[:, 0:256]
    gemm(x, 300, FIN(17), 0, nullptr, deg_o, tmp, 256, N, 300, 256);
    fill(agg, 0.f, (long long)N * 256);
    scatter_add_kernel<<<(unsigned)((E * 64 + 255) / 256), 256, 0, stream>>>(tmp, src, dst, agg, E);
    ln(agg, 256, deg_i, FIN(18), FIN(20), FIN(19), feats0 + 0, 768, 1);
    // GATv2 branch -> feats0[:, 256:512]
    gemm(x, 300, FIN(2), 0, FIN(6), nullptr, fs, 256, N, 300, 256);   // Wsrc,bsrc
    gemm(x, 300, FIN(1), 0, FIN(4), nullptr, fd, 256, N, 300, 256);   // Wdst,bdst
    fill(mmax, NEG_INF, (long long)N * 8);
    fill(denom, 0.f, (long long)N * 8);
    fill(agg, 0.f, (long long)N * 256);
    attn_logit_kernel<<<(unsigned)((E + 7) / 8), 256, 0, stream>>>(fs, fd, FIN(3), src, dst, logit, mmax, E);
    attn_expsum_kernel<<<(unsigned)((E * 8 + 255) / 256), 256, 0, stream>>>(logit, mmax, dst, wexp, denom, E);
    attn_agg_kernel<<<(unsigned)((E * 64 + 255) / 256), 256, 0, stream>>>(fs, wexp, denom, src, dst, agg, E);
    ln(agg, 256, nullptr, FIN(5), FIN(8), FIN(7), feats0 + 256, 768, 1);
    // Shortcut(x) -> feats0[:, 512:768]
    gemm(x, 300, FIN(37), 0, FIN(38), nullptr, tmp, 256, N, 300, 256);
    ln(tmp, 256, nullptr, nullptr, FIN(40), FIN(39), feats0 + 512, 768, 1);
    // SE module (nb=3): z = flat @ W + b, gate, sum -> h1
    gemm(feats0, 768, FIN(29), 1, FIN(30), nullptr, z0, 768, N, 768, 768);
    semul_gate_kernel<<<(unsigned)(((long long)N * 256 + 255) / 256), 256, 0, stream>>>(
        feats0, z0, 768, FIN(32), FIN(31), 3, h1, N);

    // ================= Layer 1 (input h1, 256) =================
    // GraphConv -> feats1[:, 0:256]
    gemm(h1, 256, FIN(21), 0, nullptr, deg_o, tmp, 256, N, 256, 256);
    fill(agg, 0.f, (long long)N * 256);
    scatter_add_kernel<<<(unsigned)((E * 64 + 255) / 256), 256, 0, stream>>>(tmp, src, dst, agg, E);
    ln(agg, 256, deg_i, FIN(22), FIN(24), FIN(23), feats1 + 0, 1024, 1);
    // GATv2 -> feats1[:, 256:512]
    gemm(h1, 256, FIN(10), 0, FIN(14), nullptr, fs, 256, N, 256, 256);  // Wsrc,bsrc
    gemm(h1, 256, FIN(9), 0, FIN(12), nullptr, fd, 256, N, 256, 256);   // Wdst,bdst
    fill(mmax, NEG_INF, (long long)N * 8);
    fill(denom, 0.f, (long long)N * 8);
    fill(agg, 0.f, (long long)N * 256);
    attn_logit_kernel<<<(unsigned)((E + 7) / 8), 256, 0, stream>>>(fs, fd, FIN(11), src, dst, logit, mmax, E);
    attn_expsum_kernel<<<(unsigned)((E * 8 + 255) / 256), 256, 0, stream>>>(logit, mmax, dst, wexp, denom, E);
    attn_agg_kernel<<<(unsigned)((E * 64 + 255) / 256), 256, 0, stream>>>(fs, wexp, denom, src, dst, agg, E);
    ln(agg, 256, nullptr, FIN(13), FIN(16), FIN(15), feats1 + 256, 1024, 1);
    // Shortcut(x) -> feats1[:, 512:768]
    gemm(x, 300, FIN(41), 0, FIN(42), nullptr, tmp, 256, N, 300, 256);
    ln(tmp, 256, nullptr, nullptr, FIN(44), FIN(43), feats1 + 512, 1024, 1);
    // Shortcut(h1) -> feats1[:, 768:1024]
    gemm(h1, 256, FIN(45), 0, FIN(46), nullptr, tmp, 256, N, 256, 256);
    ln(tmp, 256, nullptr, nullptr, FIN(48), FIN(47), feats1 + 768, 1024, 1);
    // SE module (nb=4) -> tmp
    gemm(feats1, 1024, FIN(33), 1, FIN(34), nullptr, z1, 1024, N, 1024, 1024);
    semul_gate_kernel<<<(unsigned)(((long long)N * 256 + 255) / 256), 256, 0, stream>>>(
        feats1, z1, 1024, FIN(36), FIN(35), 4, tmp, N);

    // ---- final LN + output projection ----
    ln(tmp, 256, nullptr, nullptr, FIN(26), FIN(25), hln, 256, 0);
    out_proj_kernel<<<(unsigned)(((long long)N * 23 + 255) / 256), 256, 0, stream>>>(
        hln, FIN(27), FIN(28), (float*)d_out, N);
#undef FIN
}
